// MultiHeadLinearAttentionLayer_29085518528620
// MI455X (gfx1250) — matrix operand order repaired and checked by reading, not compiled, not run on hardware
//
#include <hip/hip_runtime.h>
#include <hip/hip_bf16.h>
#include <cstdint>
#include <cstddef>

typedef __attribute__((ext_vector_type(16))) _Float16 v16h;
typedef __attribute__((ext_vector_type(8)))  float    v8f;

namespace {
constexpr int N_NODES = 50000;
constexpr int N_EDGES = 800000;
constexpr int DIM     = 64;   // IN_DIM == d == H*D
constexpr int NH      = 8;    // heads
constexpr int DH      = 8;    // dim per head
constexpr float CLAMPV = 5.0f;
// packed-weight geometry: per weight, 4 col-tiles x 2 k-blocks x 32 lanes x 16 f16
constexpr int PW_PER_W = 4 * 2 * 32 * 16;  // 4096 halves = 8 KB
}

// ---------------------------------------------------------------------------
// One-shot weight pack: W[64][64] f32 -> f16 in WMMA-B fragment order so a
// lane's fragment is 32 contiguous bytes.
// B layout (matches A): lanes 0-15 K=kb*32+{0..7,16..23} at col, lanes 16-31 +8.
// ---------------------------------------------------------------------------
__global__ __launch_bounds__(256)
void pack_weights_kernel(const float* __restrict__ W0, const float* __restrict__ W1,
                         const float* __restrict__ W2, const float* __restrict__ W3,
                         const float* __restrict__ W4, _Float16* __restrict__ out) {
  const int t = blockIdx.x * blockDim.x + threadIdx.x;
  if (t >= 5 * PW_PER_W) return;
  const int k    = t & 15;
  const int lane = (t >> 4) & 31;
  const int kb   = (t >> 9) & 1;
  const int ct   = (t >> 10) & 3;
  const int w    = t >> 12;
  const float* W = (w == 0) ? W0 : (w == 1) ? W1 : (w == 2) ? W2 : (w == 3) ? W3 : W4;
  const int row = kb * 32 + (k & 8) * 2 + ((lane >> 4) & 1) * 8 + (k & 7);
  const int col = ct * 16 + (lane & 15);
  out[t] = (_Float16)W[row * DIM + col];
}

// ---------------------------------------------------------------------------
// A fragment (16x32 f16, §7.12.2): lane<16 -> row=lane, K={kb*32+0..7,+16..23};
// lane>=16 -> row=lane-16, K offsets +8.
// ---------------------------------------------------------------------------
__device__ __forceinline__ v16h load_a_frag(const float* __restrict__ A,
                                            int row0, int kb, int lane) {
  const int half = (lane >> 4) & 1;
  const int r    = lane & 15;
  const float* p = A + (size_t)(row0 + r) * DIM + kb * 32 + half * 8;
  v16h a;
#pragma unroll
  for (int i = 0; i < 8; ++i) a[i]     = (_Float16)p[i];
#pragma unroll
  for (int i = 0; i < 8; ++i) a[8 + i] = (_Float16)p[16 + i];
  return a;
}

// C[16][64] = A-tile @ W + bias. Weight fragments are read from LDS
// (ds_load_b128 pairs) -> no register hoisting of the whole weight matrix.
// acc[ct][i] = C[row0 + (lane>=16?8:0) + i][ct*16 + (lane&15)]
__device__ __forceinline__ void gemm16x64p(const v16h a0, const v16h a1,
                                           const _Float16* pw,  // LDS
                                           const float* __restrict__ bias,
                                           int lane, v8f acc[4]) {
#pragma unroll
  for (int ct = 0; ct < 4; ++ct) {
    const float bv = bias[ct * 16 + (lane & 15)];
    v8f c;
#pragma unroll
    for (int i = 0; i < 8; ++i) c[i] = bv;
    const v16h b0 = *(const v16h*)(pw + ((ct * 2 + 0) * 32 + lane) * 16);
    const v16h b1 = *(const v16h*)(pw + ((ct * 2 + 1) * 32 + lane) * 16);
    c = __builtin_amdgcn_wmma_f32_16x16x32_f16(false, a0, false, b0, (short)0, c, false, false);
    c = __builtin_amdgcn_wmma_f32_16x16x32_f16(false, a1, false, b1, (short)0, c, false, false);
    acc[ct] = c;
  }
}

// Store WMMA C tile to row-major [*,64]; for fixed (ct,i) a half-wave writes
// 64 contiguous bytes -> coalesced store clauses.
__device__ __forceinline__ void store_tile(float* __restrict__ out, int row0,
                                           int lane, const v8f acc[4]) {
  const int rbase = row0 + ((lane >> 4) ? 8 : 0);
  const int c     = lane & 15;
#pragma unroll
  for (int ct = 0; ct < 4; ++ct)
#pragma unroll
    for (int i = 0; i < 8; ++i)
      out[(size_t)(rbase + i) * DIM + ct * 16 + c] = acc[ct][i];
}

// cooperative global -> LDS copy of packed weights (uint4 = 8 halves)
__device__ __forceinline__ void stage_weights(_Float16* dst_lds,
                                              const _Float16* __restrict__ src,
                                              int halves) {
  uint4* d = (uint4*)dst_lds;
  const uint4* s = (const uint4*)src;
  for (int i = threadIdx.x; i < halves / 8; i += blockDim.x) d[i] = s[i];
  __syncthreads();
}

// ---------------------------------------------------------------------------
// Kernel 1: node projections Q,K,V + RoGPE rotation of Q,K.
// Weights live in LDS (24 KB/block); V stored first to cut peak live VGPRs.
// ---------------------------------------------------------------------------
__global__ __launch_bounds__(256)
void node_proj_kernel(const float* __restrict__ x,
                      const float* __restrict__ angles,
                      const _Float16* __restrict__ packw,   // [Wq|Wk|Wv|We|Wew]
                      const float* __restrict__ bq, const float* __restrict__ bk,
                      const float* __restrict__ bv,
                      const float* __restrict__ Ws, const float* __restrict__ bs,
                      float* __restrict__ Qr, float* __restrict__ Kr,
                      float* __restrict__ Vo) {
  __shared__ alignas(64) _Float16 sw[3 * PW_PER_W];  // Wq|Wk|Wv
  stage_weights(sw, packw, 3 * PW_PER_W);

  const int lane   = threadIdx.x & 31;
  const int wave   = (blockIdx.x * blockDim.x + threadIdx.x) >> 5;
  const int nwaves = (gridDim.x * blockDim.x) >> 5;
  const int ntiles = N_NODES / 16;  // 3125 exact -> EXEC all ones for WMMA

  for (int t = wave; t < ntiles; t += nwaves) {
    const int row0 = t * 16;
    if (t + nwaves < ntiles)  // prefetch next tile rows (global_prefetch_b8)
      __builtin_prefetch(x + (size_t)((t + nwaves) * 16 + (lane & 15)) * DIM, 0, 1);

    const v16h a0 = load_a_frag(x, row0, 0, lane);  // shared across Q,K,V
    const v16h a1 = load_a_frag(x, row0, 1, lane);

    v8f v[4];
    gemm16x64p(a0, a1, sw + 2 * PW_PER_W, bv, lane, v);
    store_tile(Vo, row0, lane, v);                  // retire V accumulators early

    v8f q[4], k[4];
    gemm16x64p(a0, a1, sw + 0 * PW_PER_W, bq, lane, q);
    gemm16x64p(a0, a1, sw + 1 * PW_PER_W, bk, lane, k);

    const int rbase = row0 + ((lane >> 4) ? 8 : 0);
    float ang[8];
#pragma unroll
    for (int i = 0; i < 8; ++i) ang[i] = angles[rbase + i];

#pragma unroll
    for (int ct = 0; ct < 4; ++ct) {
      const int c       = ct * 16 + (lane & 15);
      const int j       = c & 7;                               // col within head
      const int srcj    = (j < 4) ? (2 * j + 1) : (2 * (j - 4));
      const int srclane = (lane & 24) | srcj;                  // keep half + head
      const float sgn   = (j < 4) ? -1.0f : 1.0f;
      const int tix     = c >> 1;
      const float wsv = Ws[tix], bsv = bs[tix];
#pragma unroll
      for (int i = 0; i < 8; ++i) {
        const float th = ang[i] * wsv + bsv;
        const float cn = cosf(th), sn = sinf(th);
        const float q3 = sgn * __shfl(q[ct][i], srclane, 32);
        const float k3 = sgn * __shfl(k[ct][i], srclane, 32);
        q[ct][i] = q[ct][i] * cn + q3 * sn;
        k[ct][i] = k[ct][i] * cn + k3 * sn;
      }
    }
    store_tile(Qr, row0, lane, q);
    store_tile(Kr, row0, lane, k);
  }
}

// ---------------------------------------------------------------------------
// Kernel 2: edge projections E,Ew + bug-compatible power rotation of E.
// et[e,j] = theta[(e*64+j) % NE] ** power[(e*64+j) / NE], power[p] = -(p>>1)/32
// E'[2i] = E[2i]*c - E[2i+1]*s ; E'[2i+1] = E[2i+1]*c + E[2i]*s  (shfl_xor 1)
// ---------------------------------------------------------------------------
__global__ __launch_bounds__(256)
void edge_proj_kernel(const float* __restrict__ ea,
                      const float* __restrict__ theta,
                      const _Float16* __restrict__ packw,
                      const float* __restrict__ be, const float* __restrict__ bew,
                      float* __restrict__ Eo, float* __restrict__ Ewo) {
  __shared__ alignas(64) _Float16 sw[2 * PW_PER_W];  // We|Wew
  stage_weights(sw, packw + 3 * PW_PER_W, 2 * PW_PER_W);

  const int lane   = threadIdx.x & 31;
  const int wave   = (blockIdx.x * blockDim.x + threadIdx.x) >> 5;
  const int nwaves = (gridDim.x * blockDim.x) >> 5;
  const int ntiles = N_EDGES / 16;  // 50000 exact

  for (int t = wave; t < ntiles; t += nwaves) {
    const int row0 = t * 16;
    if (t + nwaves < ntiles)
      __builtin_prefetch(ea + (size_t)((t + nwaves) * 16 + (lane & 15)) * DIM, 0, 1);

    const v16h a0 = load_a_frag(ea, row0, 0, lane);  // shared across E,Ew
    const v16h a1 = load_a_frag(ea, row0, 1, lane);

    v8f ew[4];
    gemm16x64p(a0, a1, sw + 1 * PW_PER_W, bew, lane, ew);
    store_tile(Ewo, row0, lane, ew);                 // retire Ew early

    v8f e[4];
    gemm16x64p(a0, a1, sw + 0 * PW_PER_W, be, lane, e);

    const int rbase = row0 + ((lane >> 4) ? 8 : 0);
#pragma unroll
    for (int ct = 0; ct < 4; ++ct) {
      const int   c   = ct * 16 + (lane & 15);
      const float sgn = (c & 1) ? 1.0f : -1.0f;  // e2[even]=-E[c+1], e2[odd]=+E[c-1]
#pragma unroll
      for (int i = 0; i < 8; ++i) {
        const unsigned f   = (unsigned)(rbase + i) * 64u + (unsigned)c;
        const unsigned p   = f / (unsigned)N_EDGES;     // strength-reduced division
        const unsigned idx = f - p * (unsigned)N_EDGES;
        const float pw  = -(float)(p >> 1) * (1.0f / 32.0f);
        const float et  = powf(theta[idx], pw);         // theta in [0.5,1.5] > 0
        const float ce  = cosf(et), se = sinf(et);
        const float prt = __shfl_xor(e[ct][i], 1, 32);
        e[ct][i] = e[ct][i] * ce + sgn * prt * se;
      }
    }
    store_tile(Eo, row0, lane, e);
  }
}

// ---------------------------------------------------------------------------
// Kernel 3 (pass A): per (edge, head) score s, wE output, softmax denominator z.
// s clamped to [-5,5] => exp(s) stable without the segment-max pass.
// ---------------------------------------------------------------------------
__global__ __launch_bounds__(256)
void edge_score_kernel(const int* __restrict__ eidx,
                       const float* __restrict__ Kr, const float* __restrict__ Qr,
                       const float* __restrict__ Eo, const float* __restrict__ Ewo,
                       const float* __restrict__ Aw,
                       float* __restrict__ s_out, float* __restrict__ z,
                       float* __restrict__ wE_out) {
  const int stride = gridDim.x * blockDim.x;
  for (int t = blockIdx.x * blockDim.x + threadIdx.x; t < N_EDGES * NH; t += stride) {
    const int e = t >> 3, h = t & 7;
    const int src = eidx[e];
    const int dst = eidx[N_EDGES + e];
    const float4* kp = (const float4*)(Kr + (size_t)src * DIM + h * DH);
    const float4* qp = (const float4*)(Qr + (size_t)dst * DIM + h * DH);
    const float4* ep = (const float4*)(Eo + (size_t)e   * DIM + h * DH);
    const float4 k0 = kp[0], k1 = kp[1], q0 = qp[0], q1 = qp[1], e0 = ep[0], e1 = ep[1];
    float sc[8];
    sc[0] = (k0.x + q0.x + e0.x) * (1.0f / 3.0f);
    sc[1] = (k0.y + q0.y + e0.y) * (1.0f / 3.0f);
    sc[2] = (k0.z + q0.z + e0.z) * (1.0f / 3.0f);
    sc[3] = (k0.w + q0.w + e0.w) * (1.0f / 3.0f);
    sc[4] = (k1.x + q1.x + e1.x) * (1.0f / 3.0f);
    sc[5] = (k1.y + q1.y + e1.y) * (1.0f / 3.0f);
    sc[6] = (k1.z + q1.z + e1.z) * (1.0f / 3.0f);
    sc[7] = (k1.w + q1.w + e1.w) * (1.0f / 3.0f);
    float s = 0.0f;
#pragma unroll
    for (int d = 0; d < DH; ++d) s += sc[d] * Aw[d * NH + h];  // Aw[D,H,1]
    s = fminf(CLAMPV, fmaxf(-CLAMPV, s));
    s_out[t] = s;

    const float4* wp = (const float4*)(Ewo + (size_t)e * DIM + h * DH);
    const float4 w0 = wp[0], w1 = wp[1];
    float4* op = (float4*)(wE_out + (size_t)e * DIM + h * DH);
    op[0] = make_float4(w0.x * s, w0.y * s, w0.z * s, w0.w * s);
    op[1] = make_float4(w1.x * s, w1.y * s, w1.z * s, w1.w * s);

    atomicAdd(&z[(size_t)dst * NH + h], __expf(s));
  }
}

// ---------------------------------------------------------------------------
// Kernel 4 (pass B): attn = exp(s)/(z+eps); scatter wV += V[src]*attn and
// rowV += e_t*attn. e_t recomputed from L2-resident tables.
// ---------------------------------------------------------------------------
__global__ __launch_bounds__(256)
void edge_aggregate_kernel(const int* __restrict__ eidx,
                           const float* __restrict__ Kr, const float* __restrict__ Qr,
                           const float* __restrict__ Eo, const float* __restrict__ Vv,
                           const float* __restrict__ s_in, const float* __restrict__ z,
                           float* __restrict__ wV_out, float* __restrict__ rowV) {
  const int stride = gridDim.x * blockDim.x;
  for (int t = blockIdx.x * blockDim.x + threadIdx.x; t < N_EDGES * NH; t += stride) {
    const int e = t >> 3, h = t & 7;
    const int src = eidx[e];
    const int dst = eidx[N_EDGES + e];
    const float attn = __expf(s_in[t]) / (z[(size_t)dst * NH + h] + 1e-16f);

    const float* kp = Kr + (size_t)src * DIM + h * DH;
    const float* qp = Qr + (size_t)dst * DIM + h * DH;
    const float* ep = Eo + (size_t)e   * DIM + h * DH;
    const float* vp = Vv + (size_t)src * DIM + h * DH;
    float* wvp = wV_out + (size_t)dst * DIM + h * DH;
    float* rvp = rowV   + (size_t)dst * DIM + h * DH;
#pragma unroll
    for (int d = 0; d < DH; ++d) {
      atomicAdd(&wvp[d], vp[d] * attn);
      const float et = (kp[d] + qp[d] + ep[d]) * (1.0f / 3.0f);
      atomicAdd(&rvp[d], et * attn);
    }
  }
}

// ---------------------------------------------------------------------------
// Kernel 5: wV[n,h,c] += sum_d rowV[n,h,d] * VeRow[d,h,c]
// ---------------------------------------------------------------------------
__global__ __launch_bounds__(256)
void finalize_kernel(const float* __restrict__ rowV,
                     const float* __restrict__ VeRow,
                     float* __restrict__ wV_out) {
  const int stride = gridDim.x * blockDim.x;
  for (int t = blockIdx.x * blockDim.x + threadIdx.x; t < N_NODES * DIM; t += stride) {
    const int n = t >> 6, rem = t & 63, h = rem >> 3, c = rem & 7;
    float acc = wV_out[t];
    const float* rv = rowV + (size_t)n * DIM + h * DH;
#pragma unroll
    for (int d = 0; d < DH; ++d) acc += rv[d] * VeRow[(size_t)(d * NH + h) * DH + c];
    wV_out[t] = acc;
  }
}

// ---------------------------------------------------------------------------
extern "C" void kernel_launch(void* const* d_in, const int* in_sizes, int n_in,
                              void* d_out, int out_size, void* d_ws, size_t ws_size,
                              hipStream_t stream) {
  const float* x    = (const float*)d_in[0];
  const float* ea   = (const float*)d_in[1];
  const float* nang = (const float*)d_in[2];
  const float* eang = (const float*)d_in[3];
  const int*   eidx = (const int*)  d_in[4];
  const float* Wq = (const float*)d_in[5],  *bq = (const float*)d_in[6];
  const float* Wk = (const float*)d_in[7],  *bk = (const float*)d_in[8];
  const float* We = (const float*)d_in[9],  *be = (const float*)d_in[10];
  const float* Wew= (const float*)d_in[11], *bew= (const float*)d_in[12];
  const float* Wv = (const float*)d_in[13], *bv = (const float*)d_in[14];
  const float* Aw    = (const float*)d_in[15];
  const float* VeRow = (const float*)d_in[16];
  const float* Ws = (const float*)d_in[17], *bs = (const float*)d_in[18];

  float* out = (float*)d_out;
  float* wV  = out;                                 // [N, 64]
  float* wE  = out + (size_t)N_NODES * DIM;         // [NE, 64]

  // workspace layout (floats; every region is 64-float aligned)
  float* ws = (float*)d_ws;
  size_t off = 0;
  float* Qr   = ws + off; off += (size_t)N_NODES * DIM;   // 3.2M
  float* Kr   = ws + off; off += (size_t)N_NODES * DIM;   // 3.2M
  float* Vo   = ws + off; off += (size_t)N_NODES * DIM;   // 3.2M
  float* Eo   = ws + off; off += (size_t)N_EDGES * DIM;   // 51.2M
  float* Ewo  = ws + off; off += (size_t)N_EDGES * DIM;   // 51.2M
  float* s_ws = ws + off; off += (size_t)N_EDGES * NH;    // 6.4M
  float* z_ws = ws + off; off += (size_t)N_NODES * NH;    // 0.4M
  float* rowV = ws + off; off += (size_t)N_NODES * DIM;   // 3.2M
  _Float16* packw = (_Float16*)(ws + off);                // 5*4096 halves = 40 KB

  // zero the accumulators (graph-capture-safe memset nodes)
  hipMemsetAsync(z_ws, 0, (size_t)N_NODES * NH  * sizeof(float), stream);
  hipMemsetAsync(rowV, 0, (size_t)N_NODES * DIM * sizeof(float), stream);
  hipMemsetAsync(wV,   0, (size_t)N_NODES * DIM * sizeof(float), stream);

  const int BT = 256;  // 8 wave32s per block

  // pack the five 64x64 weights into f16 WMMA-B fragment order
  pack_weights_kernel<<<(5 * PW_PER_W + BT - 1) / BT, BT, 0, stream>>>(
      Wq, Wk, Wv, We, Wew, packw);

  const int node_tiles = N_NODES / 16;  // 3125
  const int edge_tiles = N_EDGES / 16;  // 50000

  node_proj_kernel<<<(node_tiles + 7) / 8, BT, 0, stream>>>(
      x, nang, packw, bq, bk, bv, Ws, bs, Qr, Kr, Vo);

  edge_proj_kernel<<<(edge_tiles + 7) / 8, BT, 0, stream>>>(
      ea, eang, packw, be, bew, Eo, Ewo);

  const int ethreads = N_EDGES * NH;
  edge_score_kernel<<<(ethreads + BT - 1) / BT, BT, 0, stream>>>(
      eidx, Kr, Qr, Eo, Ewo, Aw, s_ws, z_ws, wE);

  edge_aggregate_kernel<<<(ethreads + BT - 1) / BT, BT, 0, stream>>>(
      eidx, Kr, Qr, Eo, Vo, s_ws, z_ws, wV, rowV);

  finalize_kernel<<<(N_NODES * DIM + BT - 1) / BT, BT, 0, stream>>>(
      rowV, VeRow, wV);
}
